// MultiheadAttention_31112743092735
// MI455X (gfx1250) — matrix operand order repaired and checked by reading, not compiled, not run on hardware
//
#include <hip/hip_runtime.h>
#include <hip/hip_bf16.h>

// ---------------------------------------------------------------------------
// MultiheadAttention forward for MI455X (gfx1250, wave32, WMMA).
//   pack_x   : query (L,B,E) f32 -> xpad (B, L+2, E) bf16 (zero-padded)
//   pack_w   : in_proj_weight (3E,E,3) f32 -> wA (3E, 3E) bf16, k = t*512+i
//   qkv_gemm : per-batch GEMM (1536x512, K=1536), 64x32 tile per wave,
//              8 v_wmma_f32_16x16x32_bf16 per k-step (A/B frag reuse)
//              -> Qs[b][h][t][d]*0.125, Kk[b][h][s][d], Vt[b][h][d][s] (bf16)
//   attn     : per (b, 16-row tile): S = Q K^T (WMMA) -> LDS, parallel
//              softmax (32 lanes, shfl_xor), bf16 P in LDS, O = P V (WMMA),
//              avg_weights reduced in fixed order (deterministic).
// ---------------------------------------------------------------------------

typedef __bf16 bf16_t;
typedef __attribute__((ext_vector_type(16))) __bf16 v16bf;
typedef __attribute__((ext_vector_type(8)))  __bf16 v8bf;
typedef __attribute__((ext_vector_type(4)))  __bf16 v4bf;
typedef __attribute__((ext_vector_type(8)))  float  v8f;

#define L_DIM 512
#define B_DIM 32
#define E_DIM 512
#define H_DIM 8
#define HD    64
#define O3    1536          // 3*E
#define LP    (L_DIM + 2)   // padded length

__device__ __forceinline__ v16bf cat16(v8bf lo, v8bf hi) {
  return __builtin_shufflevector(lo, hi, 0,1,2,3,4,5,6,7,8,9,10,11,12,13,14,15);
}

__device__ __forceinline__ v8f wmma_bf16(v16bf a, v16bf b, v8f c) {
  // (neg_a, A, neg_b, B, c_mod, C, reuse_a, reuse_b)
  return __builtin_amdgcn_wmma_f32_16x16x32_bf16(false, a, false, b,
                                                 (short)0, c, false, false);
}

// ---------------------------------------------------------------------------
// xpad[b][lp][e8..e8+7]: zero-pad rows lp==0 and lp==L+1.  8 elems/thread.
__global__ void pack_x_kernel(const float* __restrict__ query,
                              bf16_t* __restrict__ xpad) {
  int idx8 = (blockIdx.x * 256 + threadIdx.x) * 8;    // B*LP*E total elems
  int e  = idx8 & (E_DIM - 1);
  int lp = (idx8 >> 9) % LP;
  int b  = idx8 / (E_DIM * LP);
  v8bf o;
  if (lp > 0 && lp < LP - 1) {
    const float* src = query + ((lp - 1) * B_DIM + b) * E_DIM + e;
    float4 a = *(const float4*)(src);
    float4 c = *(const float4*)(src + 4);
    o[0] = (bf16_t)a.x; o[1] = (bf16_t)a.y; o[2] = (bf16_t)a.z; o[3] = (bf16_t)a.w;
    o[4] = (bf16_t)c.x; o[5] = (bf16_t)c.y; o[6] = (bf16_t)c.z; o[7] = (bf16_t)c.w;
  } else {
    o = (v8bf)(bf16_t)0.0f;
  }
  *(v8bf*)(xpad + idx8) = o;
}

// wA[o][t*512 + i] = w[o][i][t];  8 consecutive k per thread (same o, t).
__global__ void pack_w_kernel(const float* __restrict__ w,
                              bf16_t* __restrict__ wA) {
  int idx8 = (blockIdx.x * 256 + threadIdx.x) * 8;    // O3*O3 total elems
  int o = idx8 / O3;
  int k = idx8 - o * O3;
  int t = k >> 9;
  int i = k & 511;
  const float* src = w + (o * E_DIM + i) * 3 + t;     // stride 3 gather
  v8bf v;
  #pragma unroll
  for (int q = 0; q < 8; ++q) v[q] = (bf16_t)src[q * 3];
  *(v8bf*)(wA + idx8) = v;
}

// ---------------------------------------------------------------------------
// QKV projection GEMM: one wave computes a 64(o) x 32(l) tile for batch b.
__global__ __launch_bounds__(128)
void qkv_gemm_kernel(const bf16_t* __restrict__ wA,
                     const bf16_t* __restrict__ xpad,
                     const float*  __restrict__ bias,
                     bf16_t* __restrict__ Qs,
                     bf16_t* __restrict__ Kk,
                     bf16_t* __restrict__ Vt) {
  const int lane = threadIdx.x & 31;
  const int wave = threadIdx.x >> 5;
  const int tile = blockIdx.x * 4 + wave;             // 24*16 tiles per batch
  const int ob = tile >> 4;                           // 64-row o block
  const int lb = tile & 15;                           // 32-col l block
  const int b  = blockIdx.y;
  const int o0 = ob * 64;
  const int l0 = lb * 32;
  const int hi = lane >> 4;
  const int lo = lane & 15;

  const bf16_t* __restrict__ xb = xpad + b * LP * E_DIM;
  const bf16_t* arow[4];
  #pragma unroll
  for (int ra = 0; ra < 4; ++ra) arow[ra] = wA + (o0 + 16 * ra + lo) * O3;

  v8f acc[4][2];
  #pragma unroll
  for (int ra = 0; ra < 4; ++ra)
    #pragma unroll
    for (int c = 0; c < 2; ++c) acc[ra][c] = (v8f)0.0f;

  for (int k0 = 0; k0 < O3; k0 += 32) {
    // A fragments (16x32 bf16): lane row, 16 contiguous k at k0+16*hi
    const int offA = k0 + 16 * hi;
    v16bf A[4];
    #pragma unroll
    for (int ra = 0; ra < 4; ++ra)
      A[ra] = cat16(*(const v8bf*)(arow[ra] + offA),
                    *(const v8bf*)(arow[ra] + offA + 8));

    // B fragments (32x16 bf16): lane col, 16 contiguous k at k0+16*hi
    const int ks = k0 + 16 * hi;
    const int t  = ks >> 9;                 // conv tap
    const int i  = ks & 511;                // input channel
    v16bf Bf[2];
    #pragma unroll
    for (int c = 0; c < 2; ++c) {
      const bf16_t* brow = xb + (l0 + 16 * c + lo + t) * E_DIM + i;
      Bf[c] = cat16(*(const v8bf*)(brow), *(const v8bf*)(brow + 8));
    }

    if (k0 + 64 < O3) {
      __builtin_prefetch((const void*)(arow[0] + offA + 64), 0, 3);
      __builtin_prefetch((const void*)(xb + (l0 + lo) * E_DIM + k0 + 64), 0, 3);
    }

    #pragma unroll
    for (int ra = 0; ra < 4; ++ra)
      #pragma unroll
      for (int c = 0; c < 2; ++c)
        acc[ra][c] = wmma_bf16(A[ra], Bf[c], acc[ra][c]);
  }

  // Epilogue: C tile VGPR r -> row M = r + 8*hi, col N = lo
  const int third = o0 >> 9;          // 0=q, 1=k, 2=v (64-block never straddles)
  #pragma unroll
  for (int ra = 0; ra < 4; ++ra) {
    #pragma unroll
    for (int c = 0; c < 2; ++c) {
      const int l = l0 + 16 * c + lo;
      #pragma unroll
      for (int r = 0; r < 8; ++r) {
        int o = o0 + 16 * ra + r + 8 * hi;
        float val = acc[ra][c][r] + bias[o];
        int e = o & 511;
        int h = e >> 6;
        int d = e & 63;
        int bh = b * H_DIM + h;
        if (third == 0) {
          Qs[(bh * L_DIM + l) * HD + d] = (bf16_t)(val * 0.125f); // fold scale
        } else if (third == 1) {
          Kk[(bh * L_DIM + l) * HD + d] = (bf16_t)val;
        } else {
          Vt[(bh * HD + d) * L_DIM + l] = (bf16_t)val;            // transposed
        }
      }
    }
  }
}

// ---------------------------------------------------------------------------
// Attention: block = (b, 16-row tile), 4 waves, wave w does heads {w, w+4}.
__global__ __launch_bounds__(128)
void attn_kernel(const bf16_t* __restrict__ Qs,
                 const bf16_t* __restrict__ Kk,
                 const bf16_t* __restrict__ Vt,
                 float* __restrict__ out) {
  __shared__ float  lds_S[4][16 * L_DIM];   // per-wave raw scores (f32)
  __shared__ bf16_t lds_P[4][16 * L_DIM];   // per-wave normalized probs (bf16)
  __shared__ float  lds_avg[16 * L_DIM];    // head-averaged attention

  const int tt   = blockIdx.x;              // row tile
  const int b    = blockIdx.y;
  const int t0   = tt * 16;
  const int lane = threadIdx.x & 31;
  const int wave = threadIdx.x >> 5;
  const int hi   = lane >> 4;
  const int lo   = lane & 15;

  for (int j = threadIdx.x * 4; j < 16 * L_DIM; j += 128 * 4)
    *(float4*)&lds_avg[j] = make_float4(0.f, 0.f, 0.f, 0.f);
  __syncthreads();

  float*  __restrict__ S  = lds_S[wave];
  bf16_t* __restrict__ Pw = lds_P[wave];

  for (int g = 0; g < 2; ++g) {
    const int hh = g * 4 + wave;
    const int bh = b * H_DIM + hh;

    // ---- Q A-fragments (16 x 64, two k-steps of 32), reused for all s-tiles
    const bf16_t* qrow = Qs + (bh * L_DIM + t0 + lo) * HD;
    v16bf AQ[2];
    #pragma unroll
    for (int j = 0; j < 2; ++j) {
      int off = j * 32 + 16 * hi;
      AQ[j] = cat16(*(const v8bf*)(qrow + off),
                    *(const v8bf*)(qrow + off + 8));
    }

    // ---- scores S = (Q*scale) K^T : 32 column tiles -> LDS (f32)
    const bf16_t* kbase = Kk + bh * L_DIM * HD;
    for (int st = 0; st < 32; ++st) {
      const bf16_t* krow = kbase + (st * 16 + lo) * HD;   // B col = key pos
      v8f acc = (v8f)0.0f;
      #pragma unroll
      for (int j = 0; j < 2; ++j) {
        int off = j * 32 + 16 * hi;
        v16bf B = cat16(*(const v8bf*)(krow + off),
                        *(const v8bf*)(krow + off + 8));
        acc = wmma_bf16(AQ[j], B, acc);
      }
      #pragma unroll
      for (int r = 0; r < 8; ++r)
        S[(r + 8 * hi) * L_DIM + st * 16 + lo] = acc[r];
    }

    // ---- parallel softmax: 2 lanes per row (lane&15 = row, lane>>4 = half)
    {
      float* srow = S + (lane & 15) * L_DIM + hi * 256;
      float mx = -3.402823466e38f;
      for (int j = 0; j < 256; j += 4) {
        float4 v = *(const float4*)(srow + j);
        mx = fmaxf(mx, fmaxf(fmaxf(v.x, v.y), fmaxf(v.z, v.w)));
      }
      mx = fmaxf(mx, __shfl_xor(mx, 16, 32));
      float sum = 0.0f;
      for (int j = 0; j < 256; j += 4) {
        float4 v = *(const float4*)(srow + j);
        v.x = __expf(v.x - mx); v.y = __expf(v.y - mx);
        v.z = __expf(v.z - mx); v.w = __expf(v.w - mx);
        *(float4*)(srow + j) = v;
        sum += (v.x + v.y) + (v.z + v.w);
      }
      sum += __shfl_xor(sum, 16, 32);
      const float inv = 1.0f / sum;
      bf16_t* prow = Pw + (lane & 15) * L_DIM + hi * 256;
      for (int j = 0; j < 256; j += 4) {
        float4 v = *(const float4*)(srow + j);
        v4bf p;
        p[0] = (bf16_t)(v.x * inv); p[1] = (bf16_t)(v.y * inv);
        p[2] = (bf16_t)(v.z * inv); p[3] = (bf16_t)(v.w * inv);
        *(v4bf*)(prow + j) = p;
      }
    }

    // ---- deterministic head-average accumulation (fixed order, no atomics)
    __syncthreads();
    for (int j = threadIdx.x * 4; j < 16 * L_DIM; j += 128 * 4) {
      float4 a = *(float4*)&lds_avg[j];
      #pragma unroll
      for (int w = 0; w < 4; ++w) {
        v4bf p = *(const v4bf*)&lds_P[w][j];
        a.x += 0.125f * (float)p[0];
        a.y += 0.125f * (float)p[1];
        a.z += 0.125f * (float)p[2];
        a.w += 0.125f * (float)p[3];
      }
      *(float4*)&lds_avg[j] = a;
    }
    __syncthreads();

    // ---- O = P V : ks outer (A frag reused by 4 d-tiles)
    const bf16_t* vbase = Vt + bh * HD * L_DIM;
    v8f oacc[4];
    #pragma unroll
    for (int dt = 0; dt < 4; ++dt) oacc[dt] = (v8f)0.0f;
    for (int ks = 0; ks < 16; ++ks) {
      const bf16_t* pr = Pw + lo * L_DIM + ks * 32 + 16 * hi;
      v16bf A = cat16(*(const v8bf*)(pr), *(const v8bf*)(pr + 8));
      const int off = ks * 32 + 16 * hi;
      #pragma unroll
      for (int dt = 0; dt < 4; ++dt) {
        const bf16_t* vrow = vbase + (dt * 16 + lo) * L_DIM;
        v16bf B = cat16(*(const v8bf*)(vrow + off),
                        *(const v8bf*)(vrow + off + 8));
        oacc[dt] = wmma_bf16(A, B, oacc[dt]);
      }
    }
    #pragma unroll
    for (int dt = 0; dt < 4; ++dt) {
      #pragma unroll
      for (int r = 0; r < 8; ++r) {
        int t = t0 + r + 8 * hi;
        out[(t * B_DIM + b) * E_DIM + hh * HD + dt * 16 + lo] = oacc[dt][r];
      }
    }
  }

  // ---- store avg_weights (second output, offset by |attn_output|)
  __syncthreads();
  float* __restrict__ avg_out = out + (size_t)L_DIM * B_DIM * E_DIM
                                    + (size_t)b * L_DIM * L_DIM
                                    + (size_t)t0 * L_DIM;
  for (int j = threadIdx.x * 4; j < 16 * L_DIM; j += 128 * 4)
    *(float4*)(avg_out + j) = *(float4*)&lds_avg[j];
}

// ---------------------------------------------------------------------------
extern "C" void kernel_launch(void* const* d_in, const int* in_sizes, int n_in,
                              void* d_out, int out_size, void* d_ws, size_t ws_size,
                              hipStream_t stream) {
  const float* query = (const float*)d_in[0];   // (L, B, E)
  const float* wconv = (const float*)d_in[1];   // (3E, E, 3)
  const float* bias  = (const float*)d_in[2];   // (3E,)
  // d_in[3] attention_weight, d_in[4] bpps: unused by the reference body.
  float* out = (float*)d_out;

  char* ws = (char*)d_ws;
  // Workspace layout (bytes):
  //   xpad : B*LP*E bf16        = 16,842,752
  //   wA   : 3E*3E bf16         =  4,718,592
  //   Qs/Kk/Vt : B*H*L*hd bf16  = 16,777,216 each       total ~71.9 MB
  bf16_t* xpad = (bf16_t*)(ws);
  bf16_t* wA   = (bf16_t*)(ws + 16842752u);
  bf16_t* Qs   = (bf16_t*)(ws + 21561344u);
  bf16_t* Kk   = (bf16_t*)(ws + 38338560u);
  bf16_t* Vt   = (bf16_t*)(ws + 55115776u);

  pack_x_kernel<<<(B_DIM * LP * E_DIM) / (256 * 8), 256, 0, stream>>>(query, xpad);
  pack_w_kernel<<<(O3 * O3) / (256 * 8), 256, 0, stream>>>(wconv, wA);
  qkv_gemm_kernel<<<dim3((24 * 16) / 4, B_DIM), 128, 0, stream>>>(
      wA, xpad, bias, Qs, Kk, Vt);
  attn_kernel<<<dim3(L_DIM / 16, B_DIM), 128, 0, stream>>>(Qs, Kk, Vt, out);
}
